// CAM_29042568856108
// MI455X (gfx1250) — matrix operand order repaired and checked by reading, not compiled, not run on hardware
//
#include <hip/hip_runtime.h>
#include <hip/hip_bf16.h>
#include <math.h>

typedef __attribute__((ext_vector_type(16))) _Float16 v16h;
typedef __attribute__((ext_vector_type(8)))  float    v8f;
typedef __attribute__((ext_vector_type(4)))  float    v4f;

#define B_    8
#define C_    256
#define N_    4096
#define MTILE 128
#define KT    32
#define ITERS (N_ / KT)       // 128
#define KPAD  264             // 256 + 8 halfs pad (bank-conflict-free B-frag reads)
#define VPAD  40              // 32 + 8 halfs pad
#define PPAD  40              // 32 + 8 halfs pad
#define OPAD  132             // 128 + 4 floats pad

// LDS layout (bytes)
#define OFF_KHI 0
#define OFF_KLO (32 * KPAD * 2)                    // 16896
#define OFF_VHI (2 * 32 * KPAD * 2)                // 33792
#define OFF_P   (OFF_VHI + C_ * VPAD * 2)          // 54272
#define SMEM_SZ (OFF_P + 8 * 16 * PPAD * 2)        // 64512 bytes

__device__ __forceinline__ v16h ld16(const _Float16* p) {
    v16h r;
    ((uint4*)&r)[0] = ((const uint4*)p)[0];
    ((uint4*)&r)[1] = ((const uint4*)p)[1];
    return r;
}

__device__ __forceinline__ v8f wmma_hh(v16h a, v16h b, v8f c) {
    // D = A(16x32 f16) * B(32x16 f16) + C(16x16 f32)
    return __builtin_amdgcn_wmma_f32_16x16x32_f16(false, a, false, b,
                                                  (short)0, c, false, false);
}

// Workgroup barrier with LDS-only visibility: waits DScnt==0 (stores visible to
// the WGP) but does NOT flush LOADcnt, so register prefetch loads stay in
// flight across the barrier. __syncthreads() would emit
// s_wait_loadcnt_dscnt 0x0 and serialize the prefetch against compute.
__device__ __forceinline__ void wg_barrier_lds() {
    __asm__ volatile(
        "s_wait_dscnt 0\n\t"
        "s_barrier_signal -1\n\t"
        "s_barrier_wait -1" ::: "memory");
}

__device__ __forceinline__ float redmax16(float v) {
    v = fmaxf(v, __shfl_xor(v, 1));
    v = fmaxf(v, __shfl_xor(v, 2));
    v = fmaxf(v, __shfl_xor(v, 4));
    v = fmaxf(v, __shfl_xor(v, 8));
    return v;
}

__device__ __forceinline__ float redsum16(float v) {
    v += __shfl_xor(v, 1);
    v += __shfl_xor(v, 2);
    v += __shfl_xor(v, 4);
    v += __shfl_xor(v, 8);
    return v;
}

__global__ void __launch_bounds__(256, 1)
cam_flash_attn(const float* __restrict__ x, const float* __restrict__ gamma,
               float* __restrict__ out)
{
    __shared__ __align__(16) char smem[SMEM_SZ];
    _Float16 (*sKhi)[KPAD] = reinterpret_cast<_Float16(*)[KPAD]>(smem + OFF_KHI);
    _Float16 (*sKlo)[KPAD] = reinterpret_cast<_Float16(*)[KPAD]>(smem + OFF_KLO);
    _Float16 (*sVhi)[VPAD] = reinterpret_cast<_Float16(*)[VPAD]>(smem + OFF_VHI);

    const int tid  = threadIdx.x;
    const int wv   = tid >> 5;       // wave 0..7
    const int lane = tid & 31;
    const int ln   = lane & 15;      // position within 16-lane half
    const int hl   = lane >> 4;      // which half

    _Float16 (*sP)[PPAD] =
        reinterpret_cast<_Float16(*)[PPAD]>(smem + OFF_P + wv * 16 * PPAD * 2);

    const int b  = blockIdx.y;
    const int n0 = blockIdx.x * MTILE;
    const float* xb = x + (size_t)b * C_ * N_;

    // ---- Q fragments: 16 rows per wave, f16 hi/lo split, A-layout (16x32) ----
    const int nq = n0 + wv * 16 + ln;    // this lane's query row (M = ln)
    v16h qhi[8], qlo[8];
    #pragma unroll
    for (int kb = 0; kb < 8; ++kb) {
        #pragma unroll
        for (int i = 0; i < 16; ++i) {
            int c = kb * 32 + hl * 16 + i;
            float v = xb[(size_t)c * N_ + nq];
            _Float16 h = (_Float16)v;
            qhi[kb][i] = h;
            qlo[kb][i] = (_Float16)(v - (float)h);
        }
    }

    const v8f vzero = {0.f, 0.f, 0.f, 0.f, 0.f, 0.f, 0.f, 0.f};
    v8f acc[16];
    #pragma unroll
    for (int t = 0; t < 16; ++t) acc[t] = vzero;
    float m8[8], l8[8];
    #pragma unroll
    for (int r = 0; r < 8; ++r) { m8[r] = -INFINITY; l8[r] = 0.0f; }

    // ---- staged K/V tile loads: thread t owns channel c=t, coalesced in n ----
    const float* xrow = xb + (size_t)tid * N_;
    float4 st[8];
    #pragma unroll
    for (int j = 0; j < 8; ++j) st[j] = ((const float4*)xrow)[j];

    for (int it = 0; it < ITERS; ++it) {
        wg_barrier_lds();   // previous tile fully consumed

        // convert f32 -> f16 hi/lo and deposit both LDS views
        union { _Float16 h[32]; uint4 q[4]; } hp;
        #pragma unroll
        for (int j = 0; j < 8; ++j) {
            float vv[4] = { st[j].x, st[j].y, st[j].z, st[j].w };
            #pragma unroll
            for (int e = 0; e < 4; ++e) {
                int k = j * 4 + e;                 // key within tile
                _Float16 h  = (_Float16)vv[e];
                _Float16 lo = (_Float16)(vv[e] - (float)h);
                hp.h[k] = h;
                sKhi[k][tid] = h;                  // [key][c] view
                sKlo[k][tid] = lo;
            }
        }
        {
            uint4* pv = (uint4*)&sVhi[tid][0];     // [c][key] view, contiguous row
            pv[0] = hp.q[0]; pv[1] = hp.q[1]; pv[2] = hp.q[2]; pv[3] = hp.q[3];
        }
        wg_barrier_lds();   // tile ready

        // prefetch next tile into registers: issued at the top of the compute
        // phase so the loads complete underneath the WMMA stream; the only
        // wait is the register dependency before next iteration's LDS stores.
        if (it + 1 < ITERS) {
            const float4* pn = (const float4*)(xrow + (it + 1) * KT);
            #pragma unroll
            for (int j = 0; j < 8; ++j) st[j] = pn[j];
        }

        // ---- S = Q Kᵀ over C=256: 3 split-WMMAs per 32-wide c block ----
        v8f s0 = vzero, s1 = vzero;       // key sub-tiles [0,16) and [16,32)
        #pragma unroll
        for (int kb = 0; kb < 8; ++kb) {
            const int co = kb * 32 + hl * 16;
            v16h bh0 = ld16(&sKhi[ln][co]);
            v16h bl0 = ld16(&sKlo[ln][co]);
            s0 = wmma_hh(qhi[kb], bl0, s0);
            s0 = wmma_hh(qlo[kb], bh0, s0);
            s0 = wmma_hh(qhi[kb], bh0, s0);
            v16h bh1 = ld16(&sKhi[16 + ln][co]);
            v16h bl1 = ld16(&sKlo[16 + ln][co]);
            s1 = wmma_hh(qhi[kb], bl1, s1);
            s1 = wmma_hh(qlo[kb], bh1, s1);
            s1 = wmma_hh(qhi[kb], bh1, s1);
        }

        // ---- streaming softmax update (per-row stats live in lane halves) ----
        float alpha[8];
        #pragma unroll
        for (int r = 0; r < 8; ++r) {
            float tm = redmax16(fmaxf(s0[r], s1[r]));
            float nm = fmaxf(m8[r], tm);
            alpha[r] = __expf(m8[r] - nm);      // exp(-inf)=0 on first tile
            m8[r] = nm;
            s0[r] = __expf(s0[r] - nm);
            s1[r] = __expf(s1[r] - nm);
            l8[r] = l8[r] * alpha[r] + redsum16(s0[r] + s1[r]);
        }

        // ---- P: C-layout -> A-layout via per-wave LDS scratch ----
        #pragma unroll
        for (int r = 0; r < 8; ++r) {
            sP[r + 8 * hl][ln]      = (_Float16)s0[r];
            sP[r + 8 * hl][16 + ln] = (_Float16)s1[r];
        }
        __asm__ volatile("s_wait_dscnt 0" ::: "memory");  // wave-local DS ordering
        v16h pA;
        ((uint4*)&pA)[0] = *(const uint4*)&sP[ln][hl * 16];
        ((uint4*)&pA)[1] = *(const uint4*)&sP[ln][hl * 16 + 8];

        // ---- O = O*alpha + P·V ----
        #pragma unroll
        for (int t = 0; t < 16; ++t) {
            v8f a = acc[t];
            #pragma unroll
            for (int r = 0; r < 8; ++r) a[r] *= alpha[r];
            v16h vb = ld16(&sVhi[t * 16 + ln][hl * 16]);
            acc[t] = wmma_hh(pA, vb, a);
        }
    }

    // ---- epilogue: O/l, transpose via LDS, coalesced gamma*O + x stores ----
    float rl[8];
    #pragma unroll
    for (int r = 0; r < 8; ++r) rl[r] = 1.0f / l8[r];
    const float g = gamma[0];

    float (*sOut)[OPAD] = reinterpret_cast<float(*)[OPAD]>(smem); // aliases K region
    float* outb = out + (size_t)b * C_ * N_;

    for (int cc = 0; cc < 8; ++cc) {        // 32 channels per pass
        __syncthreads();
        #pragma unroll
        for (int sub = 0; sub < 2; ++sub) {
            int t = cc * 2 + sub;
            #pragma unroll
            for (int r = 0; r < 8; ++r)
                sOut[sub * 16 + ln][wv * 16 + r + 8 * hl] = acc[t][r] * rl[r];
        }
        __syncthreads();
        int cl = tid >> 3;                  // 0..31 channel within chunk
        int np = (tid & 7) * 16;            // 0..112 n offset
        size_t gofs = (size_t)(cc * 32 + cl) * N_ + n0 + np;
        #pragma unroll
        for (int q = 0; q < 4; ++q) {
            const v4f o  = *(const v4f*)&sOut[cl][np + q * 4];
            const v4f xi = *(const v4f*)&xb[gofs + q * 4];
            v4f res = g * o + xi;
            // write-once output: non-temporal so it doesn't evict the shared
            // K/V stream from L2
            __builtin_nontemporal_store(res, (v4f*)&outb[gofs + q * 4]);
        }
    }
}

extern "C" void kernel_launch(void* const* d_in, const int* in_sizes, int n_in,
                              void* d_out, int out_size, void* d_ws, size_t ws_size,
                              hipStream_t stream) {
    const float* x     = (const float*)d_in[0];
    const float* gamma = (const float*)d_in[1];
    float* out         = (float*)d_out;
    dim3 grid(N_ / MTILE, B_);   // 32 query tiles x 8 batches
    cam_flash_attn<<<grid, 256, 0, stream>>>(x, gamma, out);
}
